// SparseAttentionBlock_71133248356887
// MI455X (gfx1250) — 16-bit and 8-bit matrix operand order repaired and checked by reading, compiled, not run on hardware
//
#include <hip/hip_runtime.h>
#include <hip/hip_bf16.h>

typedef __attribute__((ext_vector_type(2)))  float    v2f;
typedef __attribute__((ext_vector_type(8)))  float    v8f;
typedef __attribute__((ext_vector_type(8)))  int      v8i;
typedef __attribute__((ext_vector_type(16))) _Float16 v16h;

#define T_STEPS 4
#define BATCH   8
#define NTOK    1024
#define CH      512
#define HEADS   8
#define HDIM    64
#define V_TH    1.0f
#define BN_EPS  1e-5f

__global__ __launch_bounds__(256) void gemm_bn_lif(
    const float* __restrict__ X, const float* __restrict__ W,
    const float* __restrict__ bias, const float* __restrict__ bn, int bn_layer,
    int mode, _Float16* __restrict__ outH, unsigned char* __restrict__ out8,
    float* __restrict__ outF)
{
  const int n0   = blockIdx.x * 32;
  const int col0 = blockIdx.y * 64;
  const int bz   = blockIdx.z;
  const int tid  = threadIdx.x;
  const int wv   = tid >> 5;
  const int lane = tid & 31;
  const int rsel = lane & 15;
  const int hi   = lane >> 4;
  const int ksel = hi * 2;

  __shared__ union SM {
    struct { float Xs[2][128][36]; float Wt[2][64][36]; } t;
    float Ys[128][66];
  } sm;

  const int r0 = tid >> 3;
  const int c4 = (tid & 7) * 4;
  const float* xbase[4];
  #pragma unroll
  for (int t = 0; t < 4; ++t)
    xbase[t] = X + ((size_t)((t * BATCH + bz) * NTOK) + n0 + r0) * CH + c4;
  const float* wbase[2];
  #pragma unroll
  for (int q = 0; q < 2; ++q)
    wbase[q] = W + (size_t)(col0 + r0 + q * 32) * CH + c4;

  v8f acc[4];
  #pragma unroll
  for (int j = 0; j < 4; ++j) acc[j] = (v8f){0,0,0,0,0,0,0,0};

  float4 xr[4], wr[2];
  #pragma unroll
  for (int t = 0; t < 4; ++t) xr[t] = *(const float4*)(xbase[t]);
  #pragma unroll
  for (int q = 0; q < 2; ++q) wr[q] = *(const float4*)(wbase[q]);
  #pragma unroll
  for (int t = 0; t < 4; ++t) *(float4*)&sm.t.Xs[0][t * 32 + r0][c4] = xr[t];
  #pragma unroll
  for (int q = 0; q < 2; ++q) *(float4*)&sm.t.Wt[0][r0 + q * 32][c4] = wr[q];
  __syncthreads();

  int buf = 0;
  for (int kc = 0; kc < CH; kc += 32) {
    const bool more = (kc + 32 < CH);
    if (more) {
      #pragma unroll
      for (int t = 0; t < 4; ++t) xr[t] = *(const float4*)(xbase[t] + kc + 32);
      #pragma unroll
      for (int q = 0; q < 2; ++q) wr[q] = *(const float4*)(wbase[q] + kc + 32);
    }
    #pragma unroll
    for (int k = 0; k < 32; k += 4) {
      v2f a;
      a.x = sm.t.Xs[buf][16 * wv + rsel][k + ksel];
      a.y = sm.t.Xs[buf][16 * wv + rsel][k + ksel + 1];
      v2f bf[4];
      #pragma unroll
      for (int j = 0; j < 4; ++j) {
        bf[j].x = sm.t.Wt[buf][j * 16 + rsel][k + ksel];
        bf[j].y = sm.t.Wt[buf][j * 16 + rsel][k + ksel + 1];
      }
      #pragma unroll
      for (int j = 0; j < 4; ++j)
        acc[j] = __builtin_amdgcn_wmma_f32_16x16x4_f32(
            false, a, false, bf[j], (short)0, acc[j], false, false);
    }
    if (more) {
      #pragma unroll
      for (int t = 0; t < 4; ++t) *(float4*)&sm.t.Xs[buf ^ 1][t * 32 + r0][c4] = xr[t];
      #pragma unroll
      for (int q = 0; q < 2; ++q) *(float4*)&sm.t.Wt[buf ^ 1][r0 + q * 32][c4] = wr[q];
    }
    __syncthreads();
    buf ^= 1;
  }

  {
    const int mhalf = hi * 8;
    #pragma unroll
    for (int j = 0; j < 4; ++j)
      #pragma unroll
      for (int r = 0; r < 8; ++r)
        sm.Ys[16 * wv + mhalf + r][j * 16 + rsel] = acc[j][r];
  }
  __syncthreads();

  #pragma unroll
  for (int kk = 0; kk < 8; ++kk) {
    int p   = tid + kk * 256;
    int i   = p >> 6;
    int col = p & 63;
    int c   = col0 + col;
    float g   = bn[(bn_layer * 4 + 0) * CH + c];
    float be  = bn[(bn_layer * 4 + 1) * CH + c];
    float mu  = bn[(bn_layer * 4 + 2) * CH + c];
    float var = bn[(bn_layer * 4 + 3) * CH + c];
    float sc  = rsqrtf(var + BN_EPS) * g;
    float bi  = bias[c];
    float mem = 0.0f;
    #pragma unroll
    for (int t = 0; t < T_STEPS; ++t) {
      float y = sm.Ys[t * 32 + i][col] + bi;
      y = (y - mu) * sc + be;
      mem += (y - mem) * 0.5f;
      float s = (mem >= V_TH) ? 1.0f : 0.0f;
      mem *= (1.0f - s);
      if (mode == 0) {
        int h = c >> 6, d = c & 63;
        outH[(((size_t)((t * BATCH + bz) * HEADS + h)) * NTOK + (n0 + i)) * HDIM + d] =
            (_Float16)s;
      } else if (mode == 1) {
        int h = c >> 6, d = c & 63;
        out8[(((size_t)((t * BATCH + bz) * HEADS + h)) * HDIM + d) * NTOK + (n0 + i)] =
            (unsigned char)s;
      } else {
        outF[((size_t)((t * BATCH + bz) * NTOK) + (n0 + i)) * CH + c] = s;
      }
    }
  }
}

__global__ __launch_bounds__(128) void attn_kernel(
    const _Float16* __restrict__ qf, const unsigned char* __restrict__ kT,
    const unsigned char* __restrict__ vT, float* __restrict__ yout)
{
  const int blk  = blockIdx.x;
  const int h    = blk & 7;
  const int tb   = blk >> 3;
  const size_t base8 = (size_t)blk * HDIM * NTOK;
  const size_t baseq = (size_t)blk * NTOK * HDIM;
  const int tid = threadIdx.x, wv = tid >> 5, lane = tid & 31;
  const int rsel = lane & 15, hi = lane >> 4;

  __shared__ unsigned char ks[2][64][64];
  __shared__ unsigned char vs[2][64][64];
  __shared__ _Float16 kvfT[64][72];

  v8i kvacc[4];
  #pragma unroll
  for (int j = 0; j < 4; ++j) kvacc[j] = (v8i){0,0,0,0,0,0,0,0};

  const int sd  = tid >> 1;
  const int soff = (tid & 1) * 32;
  const unsigned char* kbase = kT + base8 + (size_t)sd * NTOK + soff;
  const unsigned char* vbase = vT + base8 + (size_t)sd * NTOK + soff;

  uint4 kr[2], vr[2];
  kr[0] = *(const uint4*)(kbase);      kr[1] = *(const uint4*)(kbase + 16);
  vr[0] = *(const uint4*)(vbase);      vr[1] = *(const uint4*)(vbase + 16);
  *(uint4*)&ks[0][sd][soff] = kr[0];   *(uint4*)&ks[0][sd][soff + 16] = kr[1];
  *(uint4*)&vs[0][sd][soff] = vr[0];   *(uint4*)&vs[0][sd][soff + 16] = vr[1];
  __syncthreads();

  int buf = 0;
  for (int nn = 0; nn < NTOK; nn += 64) {
    const bool more = (nn + 64 < NTOK);
    if (more) {
      kr[0] = *(const uint4*)(kbase + nn + 64);
      kr[1] = *(const uint4*)(kbase + nn + 80);
      vr[0] = *(const uint4*)(vbase + nn + 64);
      vr[1] = *(const uint4*)(vbase + nn + 80);
    }
    const unsigned char* arow = &ks[buf][16 * wv + rsel][0];
    v8i A;
    A[0] = *(const int*)(arow + (hi ?  8 :  0));
    A[1] = *(const int*)(arow + (hi ? 12 :  4));
    A[2] = *(const int*)(arow + (hi ? 24 : 16));
    A[3] = *(const int*)(arow + (hi ? 28 : 20));
    A[4] = *(const int*)(arow + 32 + (hi ?  8 :  0));
    A[5] = *(const int*)(arow + 32 + (hi ? 12 :  4));
    A[6] = *(const int*)(arow + 32 + (hi ? 24 : 16));
    A[7] = *(const int*)(arow + 32 + (hi ? 28 : 20));
    v8i Bv[4];
    #pragma unroll
    for (int j = 0; j < 4; ++j) {
      const unsigned char* brow = &vs[buf][j * 16 + rsel][0];
      int kb = hi * 8;
      Bv[j][0] = *(const int*)(brow + kb + 0);
      Bv[j][1] = *(const int*)(brow + kb + 4);
      Bv[j][2] = *(const int*)(brow + kb + 16);
      Bv[j][3] = *(const int*)(brow + kb + 20);
      Bv[j][4] = *(const int*)(brow + 32 + kb + 0);
      Bv[j][5] = *(const int*)(brow + 32 + kb + 4);
      Bv[j][6] = *(const int*)(brow + 32 + kb + 16);
      Bv[j][7] = *(const int*)(brow + 32 + kb + 20);
    }
    #pragma unroll
    for (int j = 0; j < 4; ++j)
      kvacc[j] = __builtin_amdgcn_wmma_i32_16x16x64_iu8(
          false, A, false, Bv[j], kvacc[j], false, false);
    if (more) {
      *(uint4*)&ks[buf ^ 1][sd][soff]      = kr[0];
      *(uint4*)&ks[buf ^ 1][sd][soff + 16] = kr[1];
      *(uint4*)&vs[buf ^ 1][sd][soff]      = vr[0];
      *(uint4*)&vs[buf ^ 1][sd][soff + 16] = vr[1];
    }
    __syncthreads();
    buf ^= 1;
  }

  {
    const int mhalf = hi * 8;
    #pragma unroll
    for (int j = 0; j < 4; ++j)
      #pragma unroll
      for (int r = 0; r < 8; ++r)
        kvfT[j * 16 + rsel][16 * wv + mhalf + r] =
            (_Float16)((float)kvacc[j][r] * (1.0f / 1024.0f));
  }
  __syncthreads();

  union U16 { v16h hv; int iv[8]; };
  const int koff = hi * 8;
  U16 Bq[2][4];
  #pragma unroll
  for (int kstep = 0; kstep < 2; ++kstep) {
    const int kbase2 = kstep * 32 + hi * 16;
    #pragma unroll
    for (int j = 0; j < 4; ++j) {
      const int ecol = j * 16 + rsel;
      #pragma unroll
      for (int v = 0; v < 8; ++v)
        Bq[kstep][j].iv[v] = *(const int*)&kvfT[ecol][kbase2 + 2 * v];
    }
  }

  for (int m = 0; m < 16; ++m) {
    int nb = (wv * 16 + m) * 16;
    v8f acc2[4];
    #pragma unroll
    for (int j = 0; j < 4; ++j) acc2[j] = (v8f){0,0,0,0,0,0,0,0};
    #pragma unroll
    for (int kstep = 0; kstep < 2; ++kstep) {
      const _Float16* qrow = qf + baseq + (size_t)(nb + rsel) * HDIM + kstep * 32;
      U16 Aq;
      #pragma unroll
      for (int v = 0; v < 8; ++v) {
        int k = 2 * v + 2 * koff;
        Aq.iv[v] = *(const int*)(qrow + k);
      }
      #pragma unroll
      for (int j = 0; j < 4; ++j)
        acc2[j] = __builtin_amdgcn_wmma_f32_16x16x32_f16(
            false, Aq.hv, false, Bq[kstep][j].hv, (short)0, acc2[j], false, false);
    }
    const int hi8 = hi * 8;
    #pragma unroll
    for (int j = 0; j < 4; ++j)
      #pragma unroll
      for (int r = 0; r < 8; ++r) {
        int n = nb + r + hi8;
        yout[((size_t)tb * NTOK + n) * CH + h * HDIM + j * 16 + rsel] = acc2[j][r];
      }
  }
}

__global__ __launch_bounds__(256) void attn_lif(
    const float* __restrict__ y, float* __restrict__ s)
{
  size_t j = (size_t)blockIdx.x * 256 + threadIdx.x;
  const size_t stride = (size_t)BATCH * NTOK * CH;
  float mem = 0.0f;
  #pragma unroll
  for (int t = 0; t < T_STEPS; ++t) {
    float yt = y[t * stride + j];
    mem += (yt - mem) * 0.5f;
    float sp = (mem >= V_TH) ? 1.0f : 0.0f;
    s[t * stride + j] = sp;
    mem *= (1.0f - sp);
  }
}

extern "C" void kernel_launch(void* const* d_in, const int* in_sizes, int n_in,
                              void* d_out, int out_size, void* d_ws, size_t ws_size,
                              hipStream_t stream) {
  (void)in_sizes; (void)n_in; (void)out_size; (void)ws_size;
  const float* x   = (const float*)d_in[0];
  const float* Wq  = (const float*)d_in[1];
  const float* bq  = (const float*)d_in[2];
  const float* Wk  = (const float*)d_in[3];
  const float* bk  = (const float*)d_in[4];
  const float* Wv  = (const float*)d_in[5];
  const float* bv  = (const float*)d_in[6];
  const float* Wp  = (const float*)d_in[7];
  const float* bp  = (const float*)d_in[8];
  const float* bnp = (const float*)d_in[9];
  float* out = (float*)d_out;

  char* ws = (char*)d_ws;
  const size_t ELEM = (size_t)T_STEPS * BATCH * NTOK * CH;
  _Float16*      qf16   = (_Float16*)ws;                       ws += ELEM * 2;
  unsigned char* kT     = (unsigned char*)ws;                  ws += ELEM;
  unsigned char* vT     = (unsigned char*)ws;                  ws += ELEM;
  float*         y_attn = (float*)ws;                          ws += ELEM * 4;
  float*         spikes = (float*)ws;

  dim3 g(NTOK / 32, CH / 64, BATCH), blk(256);
  gemm_bn_lif<<<g, blk, 0, stream>>>(x, Wq, bq, bnp, 0, 0, qf16, nullptr, nullptr);
  gemm_bn_lif<<<g, blk, 0, stream>>>(x, Wk, bk, bnp, 1, 1, nullptr, kT, nullptr);
  gemm_bn_lif<<<g, blk, 0, stream>>>(x, Wv, bv, bnp, 2, 1, nullptr, vT, nullptr);
  attn_kernel<<<T_STEPS * BATCH * HEADS, 128, 0, stream>>>(qf16, kT, vT, y_attn);
  attn_lif<<<(BATCH * NTOK * CH) / 256, 256, 0, stream>>>(y_attn, spikes);
  gemm_bn_lif<<<g, blk, 0, stream>>>(spikes, Wp, bp, bnp, 3, 3, nullptr, nullptr, out);
}
